// MultiHeadAttention_v1_41102837023259
// MI455X (gfx1250) — matrix operand order repaired and checked by reading, not compiled, not run on hardware
//
#include <hip/hip_runtime.h>

// MHA for MI455X (gfx1250, wave32). All GEMMs via v_wmma_f32_16x16x32_f16.
// Compute-bound (~55 GFLOP vs ~60MB HBM traffic); f16 inputs / f32 accum.

typedef __attribute__((ext_vector_type(16))) _Float16 v16h;
typedef __attribute__((ext_vector_type(8)))  _Float16 v8h;
typedef __attribute__((ext_vector_type(8)))  float    v8f;

#define WMMA_F16(a, b, c) \
  __builtin_amdgcn_wmma_f32_16x16x32_f16(false, (a), false, (b), (short)0, (c), false, false)

// ---- constants ----
#define BATCH 8
#define SEQ   1024
#define DIN   768
#define HEADS 12
#define DH    64
#define BH    (BATCH * HEADS)          // 96

// workspace offsets in _Float16 units
#define OFF_XB 0u                       // [B*S, DIN]            6,291,456
#define OFF_WT 6291456u                 // [3, H, DH, DIN]       1,769,472
#define OFF_Q  8060928u                 // [BH, S, DH]           6,291,456
#define OFF_K  14352384u                // [BH, S, DH]           6,291,456
#define OFF_VT 20643840u                // [BH, DH, S]           6,291,456

// A fragment 16x32 f16, row-major source (row = lane&15).
// lane<16: K k0..k0+7 (v0-3), k0+16..k0+23 (v4-7); lane>=16: +8 shift.
__device__ __forceinline__ v16h load_a_frag(const _Float16* base, int ld,
                                            int row, int k0, int halfsel) {
  const _Float16* p0 = base + (size_t)row * ld + k0 + 8 * halfsel;
  v16h a;
  *(v8h*)&a       = *(const v8h*)p0;
  *((v8h*)&a + 1) = *(const v8h*)(p0 + 16);
  return a;
}

// B fragment 32x16 f16 from an n-major (transposed) source: Bt[n][k].
// lane<16: K k0..k0+7, k0+16..k0+23; lane>=16: +8 shift (matches A-frag).
__device__ __forceinline__ v16h load_b_frag(const _Float16* base, int ld,
                                            int n, int k0, int halfsel) {
  const _Float16* p = base + (size_t)n * ld + k0 + 8 * halfsel;
  v16h b;
  *(v8h*)&b       = *(const v8h*)p;
  *((v8h*)&b + 1) = *(const v8h*)(p + 16);
  return b;
}

// ---------------- convert kernels ----------------
__global__ void mha_cvt_x(const float* __restrict__ x, _Float16* __restrict__ xb, int n) {
  int i = blockIdx.x * blockDim.x + threadIdx.x;
  if (i < n) xb[i] = (_Float16)x[i];
}

// Wt[mat][h][e][d] = W_mat[h][d][e]  (transpose so B-frags are contiguous)
__global__ void mha_cvt_w(const float* __restrict__ Wq, const float* __restrict__ Wk,
                          const float* __restrict__ Wv, _Float16* __restrict__ Wt) {
  int i = blockIdx.x * blockDim.x + threadIdx.x;
  if (i >= 3 * HEADS * DH * DIN) return;
  int d   = i % DIN;
  int e   = (i / DIN) % DH;
  int h   = (i / (DIN * DH)) % HEADS;
  int mat = i / (DIN * DH * HEADS);
  const float* W = (mat == 0) ? Wq : (mat == 1 ? Wk : Wv);
  Wt[i] = (_Float16)W[((size_t)h * DIN + d) * DH + e];
}

// ---------------- projection GEMM ----------------
// one wave: 32 seq rows x 64 head-dim for one of {Q,K,V} of one (b,h).
// 8 WMMAs per k-step off 2 A-frags + 4 B-frags (batched loads -> one wait
// amortized over 8 WMMAs; B traffic halved vs M=16).
__global__ void __launch_bounds__(256) mha_proj(const _Float16* __restrict__ xb,
                                                const _Float16* __restrict__ Wt,
                                                _Float16* __restrict__ Q,
                                                _Float16* __restrict__ K,
                                                _Float16* __restrict__ Vt) {
  int unit = blockIdx.x * 8 + (threadIdx.x >> 5);   // 0..9215
  int lane = threadIdx.x & 31;
  int halfsel = lane >> 4, lrow = lane & 15;

  int st  = unit & 31;            // seq tile of 32 rows
  int mh  = unit >> 5;            // 0..287
  int mat = mh % 3;
  int bh  = mh / 3;
  int b   = bh / HEADS, h = bh % HEADS;
  int s0  = st * 32;

  const _Float16* A  = xb + (size_t)b * SEQ * DIN;
  const _Float16* Bt = Wt + ((size_t)mat * HEADS + h) * DH * DIN;

  v8f acc[2][4] = {};
#pragma unroll 2
  for (int kc = 0; kc < DIN / 32; ++kc) {
    int k0 = kc * 32;
    v16h a0 = load_a_frag(A, DIN, s0 + lrow, k0, halfsel);
    v16h a1 = load_a_frag(A, DIN, s0 + 16 + lrow, k0, halfsel);
    v16h bf[4];
#pragma unroll
    for (int j = 0; j < 4; ++j)
      bf[j] = load_b_frag(Bt, DIN, j * 16 + lrow, k0, halfsel);
#pragma unroll
    for (int j = 0; j < 4; ++j) {
      acc[0][j] = WMMA_F16(a0, bf[j], acc[0][j]);
      acc[1][j] = WMMA_F16(a1, bf[j], acc[1][j]);
    }
  }

  if (mat == 0) {  // fold 1/sqrt(DH) into Q once, instead of per-score
#pragma unroll
    for (int g = 0; g < 2; ++g)
#pragma unroll
      for (int j = 0; j < 4; ++j)
#pragma unroll
        for (int r = 0; r < 8; ++r)
          acc[g][j][r] *= 0.125f;
  }

  if (mat < 2) {  // Q, K stored row-major [s][dh]
    _Float16* D = (mat == 0 ? Q : K) + (size_t)bh * SEQ * DH;
#pragma unroll
    for (int g = 0; g < 2; ++g)
#pragma unroll
      for (int j = 0; j < 4; ++j)
#pragma unroll
        for (int r = 0; r < 8; ++r)
          D[(size_t)(s0 + g * 16 + r + 8 * halfsel) * DH + j * 16 + lrow] =
              (_Float16)acc[g][j][r];
  } else {        // V stored transposed [dh][s] for PV B-fragments
    _Float16* D = Vt + (size_t)bh * DH * SEQ;
#pragma unroll
    for (int g = 0; g < 2; ++g)
#pragma unroll
      for (int j = 0; j < 4; ++j)
#pragma unroll
        for (int r = 0; r < 8; ++r)
          D[(size_t)(j * 16 + lrow) * SEQ + (s0 + g * 16 + r + 8 * halfsel)] =
              (_Float16)acc[g][j][r];
  }
}

// ---------------- causal flash attention ----------------
// one wave: one 16-row q-tile of one (b,h); kv blocks of 32
__global__ void __launch_bounds__(256) mha_attn(const _Float16* __restrict__ Q,
                                                const _Float16* __restrict__ K,
                                                const _Float16* __restrict__ Vt,
                                                float* __restrict__ out) {
  __shared__ _Float16 smem[8 * 16 * 32];  // per-wave 16x32 P staging tile
  int waveId = threadIdx.x >> 5;
  int lane   = threadIdx.x & 31;
  int halfsel = lane >> 4, lrow = lane & 15;

  int unit = blockIdx.x * 8 + waveId;     // 0..6143
  int qt = unit & 63;
  int bh = unit >> 6;
  int b = bh / HEADS, h = bh % HEADS;
  int q0 = qt * 16;

  _Float16* pl = smem + waveId * (16 * 32);
  const _Float16* Qb = Q  + (size_t)bh * SEQ * DH;
  const _Float16* Kb = K  + (size_t)bh * SEQ * DH;
  const _Float16* Vb = Vt + (size_t)bh * DH * SEQ;

  v16h qf0 = load_a_frag(Qb, DH, q0 + lrow, 0, halfsel);   // Q pre-scaled
  v16h qf1 = load_a_frag(Qb, DH, q0 + lrow, 32, halfsel);

  v8f acc[4] = {};
  float m[8], l[8];
#pragma unroll
  for (int r = 0; r < 8; ++r) { m[r] = -1e30f; l[r] = 0.0f; }

  for (int kv0 = 0; kv0 <= q0 + 15; kv0 += 32) {
    // prefetch next K block rows toward WGP while this block computes
    if (kv0 + 32 <= q0 + 15)
      __builtin_prefetch(Kb + (size_t)(kv0 + 32 + lane) * DH, 0, 3);

    // ---- scores: 16x32 block (two n-halves), K-dim 64 = 2 WMMA steps ----
    // diagonal handling is wave-uniform: interior blocks skip masking.
    const bool diag = (kv0 + 31 > q0);          // block touches the diagonal
    v8f s[2];
#pragma unroll
    for (int t = 0; t < 2; ++t) {
      if (diag && (kv0 + t * 16 > q0 + 15)) {   // half-tile fully above diag
#pragma unroll
        for (int r = 0; r < 8; ++r) s[t][r] = -1e30f;
        continue;
      }
      v16h bk0 = load_b_frag(Kb, DH, kv0 + t * 16 + lrow, 0, halfsel);
      v16h bk1 = load_b_frag(Kb, DH, kv0 + t * 16 + lrow, 32, halfsel);
      v8f z = {};
      z = WMMA_F16(qf0, bk0, z);
      z = WMMA_F16(qf1, bk1, z);
      if (diag) {
        int c = kv0 + t * 16 + lrow;
#pragma unroll
        for (int r = 0; r < 8; ++r)
          s[t][r] = (c <= q0 + r + 8 * halfsel) ? z[r] : -1e30f;
      } else {
        s[t] = z;
      }
    }

    // ---- online softmax (row-reduce across 16-lane half via shfl_xor) ----
    float alpha[8];
#pragma unroll
    for (int r = 0; r < 8; ++r) {
      float rmax = fmaxf(s[0][r], s[1][r]);
      rmax = fmaxf(rmax, __shfl_xor(rmax, 8, 32));
      rmax = fmaxf(rmax, __shfl_xor(rmax, 4, 32));
      rmax = fmaxf(rmax, __shfl_xor(rmax, 2, 32));
      rmax = fmaxf(rmax, __shfl_xor(rmax, 1, 32));
      float mn = fmaxf(m[r], rmax);
      alpha[r] = __expf(m[r] - mn);
      float p0 = __expf(s[0][r] - mn);
      float p1 = __expf(s[1][r] - mn);
      s[0][r] = p0; s[1][r] = p1;
      float rs = p0 + p1;
      rs += __shfl_xor(rs, 8, 32);
      rs += __shfl_xor(rs, 4, 32);
      rs += __shfl_xor(rs, 2, 32);
      rs += __shfl_xor(rs, 1, 32);
      l[r] = l[r] * alpha[r] + rs;
      m[r] = mn;
    }
#pragma unroll
    for (int j = 0; j < 4; ++j)
#pragma unroll
      for (int r = 0; r < 8; ++r)
        acc[j][r] *= alpha[r];

    // ---- P: C-layout -> A-layout via wave-private LDS tile ----
#pragma unroll
    for (int t = 0; t < 2; ++t)
#pragma unroll
      for (int r = 0; r < 8; ++r)
        pl[(r + 8 * halfsel) * 32 + t * 16 + lrow] = (_Float16)s[t][r];
    v16h pa = load_a_frag(pl, 32, lrow, 0, halfsel);  // same-wave DS: in-order

    // ---- ctx += P * V (Vt gives n-major B-frags): batch loads, then WMMAs
    v16h bv[4];
#pragma unroll
    for (int j = 0; j < 4; ++j)
      bv[j] = load_b_frag(Vb, SEQ, j * 16 + lrow, kv0, halfsel);
#pragma unroll
    for (int j = 0; j < 4; ++j)
      acc[j] = WMMA_F16(pa, bv[j], acc[j]);
  }

  // ---- normalize and write f32 output [B,S,H*DH] ----
  float* O = out + ((size_t)b * SEQ + q0) * (HEADS * DH) + h * DH;
#pragma unroll
  for (int r = 0; r < 8; ++r) {
    float inv = 1.0f / l[r];
#pragma unroll
    for (int j = 0; j < 4; ++j)
      O[(size_t)(r + 8 * halfsel) * (HEADS * DH) + j * 16 + lrow] = acc[j][r] * inv;
  }
}

extern "C" void kernel_launch(void* const* d_in, const int* in_sizes, int n_in,
                              void* d_out, int out_size, void* d_ws, size_t ws_size,
                              hipStream_t stream) {
  const float* x  = (const float*)d_in[0];
  const float* Wq = (const float*)d_in[1];
  const float* Wk = (const float*)d_in[2];
  const float* Wv = (const float*)d_in[3];
  float* out = (float*)d_out;

  _Float16* ws = (_Float16*)d_ws;
  _Float16* xb = ws + OFF_XB;
  _Float16* Wt = ws + OFF_WT;
  _Float16* Qh = ws + OFF_Q;
  _Float16* Kh = ws + OFF_K;
  _Float16* Vt = ws + OFF_VT;

  const int nx = BATCH * SEQ * DIN;          // 6,291,456
  const int nw = 3 * HEADS * DH * DIN;       // 1,769,472
  mha_cvt_x<<<(nx + 255) / 256, 256, 0, stream>>>(x, xb, nx);
  mha_cvt_w<<<(nw + 255) / 256, 256, 0, stream>>>(Wq, Wk, Wv, Wt);

  // 96 bh * 3 mats * 32 s-tiles = 9216 wave-units / 8 waves per block
  mha_proj<<<1152, 256, 0, stream>>>(xb, Wt, Qh, Kh, Vt);

  // 96 bh * 64 q-tiles = 6144 wave-units / 8 waves per block
  mha_attn<<<768, 256, 0, stream>>>(Qh, Kh, Vt, out);
}
